// NeuralCDETransformer_71897752535683
// MI455X (gfx1250) — matrix operand order repaired and checked by reading, not compiled, not run on hardware
//
#include <hip/hip_runtime.h>

// ---------------------------------------------------------------------------
// Types for CDNA5 WMMA (wave32): bf16 A/B fragments (16 halves = 8 VGPRs),
// f32 C/D accumulator (8 VGPRs).
// ---------------------------------------------------------------------------
typedef __attribute__((ext_vector_type(16))) __bf16 v16bf;
typedef __attribute__((ext_vector_type(8)))  float  v8f;

constexpr int kC = 32, kH = 128, kO = 32, kNH = 8, kL = 4, kF = 2048;
constexpr int kB = 256, kTX = 127, kTT = 128;

// ---------------------------------------------------------------------------
// Fragment loaders matching the CDNA5 ISA layouts (CDNA5 ISA §7.12.2).
// A 16x32 (MxK), 16-bit: lane<16 holds row M=lane, K = k0+[0..7] then k0+[16..23];
//                        lane>=16 holds row M=lane-16, K = k0+[8..15], k0+[24..31].
// B 32x16 (KxN), 16-bit: lane<16 holds col N=lane, K = k0+[0..7] then k0+[16..23];
//                        lane>=16 holds col N=lane-16, K = k0+[8..15], k0+[24..31].
// ---------------------------------------------------------------------------
__device__ __forceinline__ v16bf frag_a(const __bf16* A, int lda, int k0) {
  const int lane = threadIdx.x & 31;
  const __bf16* p = A + (size_t)(lane & 15) * lda + k0 + ((lane >> 4) << 3);
  v16bf r;
  ((float4*)&r)[0] = *(const float4*)(p);        // 8 halves
  ((float4*)&r)[1] = *(const float4*)(p + 16);   // 8 halves
  return r;
}

__device__ __forceinline__ v16bf frag_b(const __bf16* W, int ldw, int n0, int k0) {
  const int lane = threadIdx.x & 31;
  const __bf16* p = W + (size_t)(n0 + (lane & 15)) * ldw + k0 + ((lane >> 4) << 3);
  v16bf r;
  ((float4*)&r)[0] = *(const float4*)(p);        // 8 halves
  ((float4*)&r)[1] = *(const float4*)(p + 16);   // 8 halves
  return r;
}

__device__ __forceinline__ v8f wmma_bf(v16bf a, v16bf b, v8f c) {
  return __builtin_amdgcn_wmma_f32_16x16x32_bf16(false, a, false, b, (short)0, c,
                                                 false, false);
}

// ---------------------------------------------------------------------------
// fp32 -> bf16 conversion
// ---------------------------------------------------------------------------
__global__ void cvt_kernel(const float* __restrict__ in, __bf16* __restrict__ out, int n) {
  int i = blockIdx.x * 256 + threadIdx.x;
  if (i < n) out[i] = (__bf16)in[i];
}

// ---------------------------------------------------------------------------
// Register-blocked WMMA GEMM for N==128 (all hidden-dim projections):
// C[M,128] = A[M,K] @ W[128,K]^T + bias.
// Block = 8 waves; wave (w) computes a 16x64 tile: m = blk*64 + (w>>1)*16,
// n = (w&1)*64. The A-fragment is reused across 4 B-tiles -> 1.25 vmem
// fragment loads per WMMA instead of 2.
// ---------------------------------------------------------------------------
__global__ __launch_bounds__(256) void gemm128_kernel(
    const __bf16* __restrict__ A, const __bf16* __restrict__ W,
    const float* __restrict__ bias, __bf16* __restrict__ Cb, int K) {
  const int lane = threadIdx.x & 31, wave = threadIdx.x >> 5;
  const int m0 = blockIdx.x * 64 + (wave >> 1) * 16;
  const int n0 = (wave & 1) * 64;
  const __bf16* Abase = A + (size_t)m0 * K;
  v8f c0 = {}, c1 = {}, c2 = {}, c3 = {};
  for (int k0 = 0; k0 < K; k0 += 32) {
    __builtin_prefetch(W + (size_t)(n0 + (lane & 15)) * K + k0 + 64, 0, 1);
    v16bf a  = frag_a(Abase, K, k0);
    c0 = wmma_bf(a, frag_b(W, K, n0,      k0), c0);
    c1 = wmma_bf(a, frag_b(W, K, n0 + 16, k0), c1);
    c2 = wmma_bf(a, frag_b(W, K, n0 + 32, k0), c2);
    c3 = wmma_bf(a, frag_b(W, K, n0 + 48, k0), c3);
  }
  const int row = m0 + ((lane >> 4) << 3);
  const int cl = lane & 15;
  auto emit = [&](v8f c, int nt) {
    const int col = n0 + nt + cl;
    const float bvv = bias[col];
#pragma unroll
    for (int i = 0; i < 8; ++i)
      Cb[(size_t)(row + i) * kH + col] = (__bf16)(c[i] + bvv);
  };
  emit(c0, 0); emit(c1, 16); emit(c2, 32); emit(c3, 48);
}

// ---------------------------------------------------------------------------
// Generic WMMA GEMM (used for the final N=32 projection):
// one wave = one 16x16 tile; grid = (N/(16*waves), M/16).
// ---------------------------------------------------------------------------
__global__ void gemm_kernel(const __bf16* __restrict__ A, const __bf16* __restrict__ W,
                            const float* __restrict__ bias, __bf16* __restrict__ Cb,
                            float* __restrict__ Cf, int N, int K) {
  const int lane = threadIdx.x & 31, wave = threadIdx.x >> 5;
  const int waves = blockDim.x >> 5;
  const int n0 = (blockIdx.x * waves + wave) * 16;
  const int m0 = blockIdx.y * 16;
  const __bf16* Abase = A + (size_t)m0 * K;
  v8f c = {};
  for (int k0 = 0; k0 < K; k0 += 32)
    c = wmma_bf(frag_a(Abase, K, k0), frag_b(W, K, n0, k0), c);
  const int col = n0 + (lane & 15);
  const int row = m0 + ((lane >> 4) << 3);
  const float bvv = bias ? bias[col] : 0.f;
#pragma unroll
  for (int i = 0; i < 8; ++i) {
    float v = c[i] + bvv;
    if (Cb) Cb[(size_t)(row + i) * N + col] = (__bf16)v;
    else    Cf[(size_t)(row + i) * N + col] = v;
  }
}

// ---------------------------------------------------------------------------
// Neural CDE scan. One persistent workgroup (8 waves) owns 16 batch samples,
// runs all 126 RK4 steps with state in LDS. The W4 (128->4096) GEMM is fused
// with the einsum('bhc,bc->bh') contraction via cross-lane reduction so the
// 16x4096 intermediate never materializes; the A-fragment is shared across
// both 16-col halves of each output feature.
// ---------------------------------------------------------------------------
__global__ __launch_bounds__(256) void cde_kernel(
    const float* __restrict__ coeffs,
    const __bf16* __restrict__ Wi, const float* __restrict__ bi,
    const __bf16* __restrict__ W1, const float* __restrict__ b1,
    const __bf16* __restrict__ W2, const float* __restrict__ b2,
    const __bf16* __restrict__ W3, const float* __restrict__ b3,
    const __bf16* __restrict__ W4, const float* __restrict__ b4,
    __bf16* __restrict__ mem) {
  __shared__ float  z[16][kH];
  __shared__ float  acc[16][kH];
  __shared__ float  kst[16][kH];
  __shared__ float  dx[16][kC];
  __shared__ __align__(16) __bf16 zt[16][kH];
  __shared__ __align__(16) __bf16 hb[16][kH];

  const int tid = threadIdx.x;
  const int lane = tid & 31;
  const int wave = tid >> 5;
  const int b0 = blockIdx.x * 16;

  // 16x128 = A(16x128) @ Wg(128x128)^T, relu, bf16 into Dst. Wave w -> cols w*16..
  auto gemm_lds = [&](const __bf16* Asrc, const __bf16* Wg, const float* bias,
                      __bf16* Dst) {
    const int n0 = wave * 16;
    v8f c = {};
#pragma unroll
    for (int k0 = 0; k0 < kH; k0 += 32)
      c = wmma_bf(frag_a(Asrc, kH, k0), frag_b(Wg, kH, n0, k0), c);
    const int col = n0 + (lane & 15);
    const int row = (lane >> 4) << 3;
    const float bvv = bias[col];
#pragma unroll
    for (int i = 0; i < 8; ++i) {
      float v = c[i] + bvv;
      Dst[(size_t)(row + i) * kH + col] = (__bf16)(v > 0.f ? v : 0.f);
    }
  };

  // m = tanh(A @ W4^T + b4) [16,4096]; g[b,h] = sum_c m[b,h*32+c]*dx[b,c] -> kst
  auto w4_contract = [&](const __bf16* Asrc) {
#pragma unroll 1
    for (int i2 = 0; i2 < 16; ++i2) {
      const int hh = wave + 8 * i2;   // this wave's output feature
      const int n0 = hh * 32;
      v8f c0 = {}, c1 = {};
#pragma unroll
      for (int k0 = 0; k0 < kH; k0 += 32) {
        v16bf a = frag_a(Asrc, kH, k0);        // shared across both halves
        c0 = wmma_bf(a, frag_b(W4, kH, n0,      k0), c0);
        c1 = wmma_bf(a, frag_b(W4, kH, n0 + 16, k0), c1);
      }
      const int rb = (lane >> 4) << 3;         // batch-row base for this lane
      const int cl = lane & 15;                // this lane's c index (mod 16)
      const float bv0 = b4[n0 + cl];
      const float bv1 = b4[n0 + 16 + cl];
      float g8[8];
#pragma unroll
      for (int i = 0; i < 8; ++i)
        g8[i] = tanhf(c0[i] + bv0) * dx[rb + i][cl]
              + tanhf(c1[i] + bv1) * dx[rb + i][16 + cl];
      // reduce over the 16 lanes (= 16 c-values) of each half-group
#pragma unroll
      for (int i = 0; i < 8; ++i) {
        float s = g8[i];
        s += __shfl_xor(s, 1, 32);
        s += __shfl_xor(s, 2, 32);
        s += __shfl_xor(s, 4, 32);
        s += __shfl_xor(s, 8, 32);
        g8[i] = s;
      }
      if ((lane & 15) == 0) {
#pragma unroll
        for (int i = 0; i < 8; ++i) kst[rb + i][hh] = g8[i];
      }
    }
  };

  auto mlp = [&]() {  // zt (bf16 z-tilde) -> kst (f32 g)
    gemm_lds(&zt[0][0], W1, b1, &hb[0][0]); __syncthreads();
    gemm_lds(&hb[0][0], W2, b2, &zt[0][0]); __syncthreads();
    gemm_lds(&zt[0][0], W3, b3, &hb[0][0]); __syncthreads();
    w4_contract(&hb[0][0]);                 __syncthreads();
  };

  // ---- z0 = coeffs[:,0,:] @ Wi^T + bi
  for (int i = tid; i < 16 * kC; i += 256) {
    int b = i >> 5, cc = i & 31;
    zt[b][cc] = (__bf16)coeffs[((size_t)(b0 + b) * kTX) * kC + cc];
  }
  __syncthreads();
  {
    const int n0 = wave * 16;
    v8f c = {};
    c = wmma_bf(frag_a(&zt[0][0], kH, 0), frag_b(Wi, kC, n0, 0), c);
    const int col = n0 + (lane & 15);
    const int row = (lane >> 4) << 3;
#pragma unroll
    for (int i = 0; i < 8; ++i) z[row + i][col] = c[i] + bi[col];
  }
  __syncthreads();
  for (int i = tid; i < 16 * kH; i += 256) {
    int b = i >> 7, hh = i & 127;
    float v = z[b][hh];
    mem[((size_t)(b0 + b) * kTX) * kH + hh] = (__bf16)v;
    zt[b][hh] = (__bf16)v;
  }
  __syncthreads();

  // ---- RK4 scan over 126 unit steps
  for (int t = 0; t < kTX - 1; ++t) {
    for (int i = tid; i < 16 * kC; i += 256) {
      int b = i >> 5, cc = i & 31;
      size_t base = ((size_t)(b0 + b) * kTX + t) * kC + cc;
      dx[b][cc] = coeffs[base + kC] - coeffs[base];
    }
    __syncthreads();
    mlp();  // k1
    for (int i = tid; i < 16 * kH; i += 256) {
      int b = i >> 7, hh = i & 127;
      float k1 = kst[b][hh];
      acc[b][hh] = z[b][hh] + k1 * (1.f / 6.f);
      zt[b][hh] = (__bf16)(z[b][hh] + 0.5f * k1);
    }
    __syncthreads();
    mlp();  // k2
    for (int i = tid; i < 16 * kH; i += 256) {
      int b = i >> 7, hh = i & 127;
      float k2 = kst[b][hh];
      acc[b][hh] += k2 * (1.f / 3.f);
      zt[b][hh] = (__bf16)(z[b][hh] + 0.5f * k2);
    }
    __syncthreads();
    mlp();  // k3
    for (int i = tid; i < 16 * kH; i += 256) {
      int b = i >> 7, hh = i & 127;
      float k3 = kst[b][hh];
      acc[b][hh] += k3 * (1.f / 3.f);
      zt[b][hh] = (__bf16)(z[b][hh] + k3);
    }
    __syncthreads();
    mlp();  // k4
    for (int i = tid; i < 16 * kH; i += 256) {
      int b = i >> 7, hh = i & 127;
      float zn = acc[b][hh] + kst[b][hh] * (1.f / 6.f);
      z[b][hh] = zn;
      zt[b][hh] = (__bf16)zn;
      mem[((size_t)(b0 + b) * kTX + (t + 1)) * kH + hh] = (__bf16)zn;
    }
    __syncthreads();
  }
}

// ---------------------------------------------------------------------------
// Streaming-softmax attention, one block per (batch, head); head_dim = 16.
// K/V staged into LDS with gfx1250 async loads (ASYNCcnt path): global ->
// LDS directly, no VGPR round-trip, then s_wait_asynccnt + barrier.
// ---------------------------------------------------------------------------
__global__ __launch_bounds__(128) void attn_kernel(
    const __bf16* __restrict__ Q, const __bf16* __restrict__ K,
    const __bf16* __restrict__ V, __bf16* __restrict__ O, int Sk) {
  __shared__ __align__(16) __bf16 KsB[128][16];
  __shared__ __align__(16) __bf16 VsB[128][16];
  __shared__ float Ks[128][16];
  __shared__ float Vs[128][16];
  const int b = blockIdx.x >> 3;
  const int h = blockIdx.x & 7;
  const int tid = threadIdx.x;

  // async copy: each 16B chunk i covers row k = i>>1, half hf = i&1
  const unsigned ldsK = (unsigned)(uintptr_t)(&KsB[0][0]);
  const unsigned ldsV = (unsigned)(uintptr_t)(&VsB[0][0]);
  for (int i = tid; i < Sk * 2; i += 128) {
    const int k = i >> 1, hf = i & 1;
    const size_t gidx = ((size_t)(b * Sk + k)) * kH + h * 16 + hf * 8;
    const __bf16* gk = K + gidx;
    const __bf16* gv = V + gidx;
    const unsigned dk = ldsK + (unsigned)i * 16u;
    const unsigned dv = ldsV + (unsigned)i * 16u;
    asm volatile("global_load_async_to_lds_b128 %0, %1, off"
                 :: "v"(dk), "v"(gk) : "memory");
    asm volatile("global_load_async_to_lds_b128 %0, %1, off"
                 :: "v"(dv), "v"(gv) : "memory");
  }
  asm volatile("s_wait_asynccnt 0x0" ::: "memory");
  __syncthreads();
  // one-time bf16 -> f32 widening pass
  for (int i = tid; i < Sk * 16; i += 128) {
    int k = i >> 4, d = i & 15;
    Ks[k][d] = (float)KsB[k][d];
    Vs[k][d] = (float)VsB[k][d];
  }
  __syncthreads();

  const int q = tid;  // Sq == 128 == blockDim.x
  float qr[16];
#pragma unroll
  for (int d = 0; d < 16; ++d)
    qr[d] = (float)Q[((size_t)(b * kTT + q)) * kH + h * 16 + d] * 0.25f;  // 1/sqrt(16)
  float mx = -1e30f, sum = 0.f, accv[16];
#pragma unroll
  for (int d = 0; d < 16; ++d) accv[d] = 0.f;
  for (int k = 0; k < Sk; ++k) {
    float s = 0.f;
#pragma unroll
    for (int d = 0; d < 16; ++d) s += qr[d] * Ks[k][d];
    float nm = s > mx ? s : mx;
    float corr = __expf(mx - nm);
    float w = __expf(s - nm);
    sum = sum * corr + w;
#pragma unroll
    for (int d = 0; d < 16; ++d) accv[d] = accv[d] * corr + w * Vs[k][d];
    mx = nm;
  }
  float inv = 1.f / sum;
#pragma unroll
  for (int d = 0; d < 16; ++d)
    O[((size_t)(b * kTT + q)) * kH + h * 16 + d] = (__bf16)(accv[d] * inv);
}

// ---------------------------------------------------------------------------
// Post-norm residual LayerNorm: x = LN(x + r)*g + bt. One wave per 128-row.
// ---------------------------------------------------------------------------
__global__ __launch_bounds__(256) void ln_kernel(__bf16* __restrict__ x,
                                                 const __bf16* __restrict__ r,
                                                 const float* __restrict__ g,
                                                 const float* __restrict__ bt) {
  const int lane = threadIdx.x & 31, wave = threadIdx.x >> 5;
  const size_t row = (size_t)blockIdx.x * 8 + wave;
  __bf16* xr = x + row * kH;
  const __bf16* rr = r + row * kH;
  float v[4], s = 0.f;
#pragma unroll
  for (int i = 0; i < 4; ++i) {
    int c = lane * 4 + i;
    v[i] = (float)xr[c] + (float)rr[c];
    s += v[i];
  }
  for (int m = 16; m; m >>= 1) s += __shfl_xor(s, m, 32);
  const float mean = s * (1.f / kH);
  float qv = 0.f;
#pragma unroll
  for (int i = 0; i < 4; ++i) { float d = v[i] - mean; qv += d * d; }
  for (int m = 16; m; m >>= 1) qv += __shfl_xor(qv, m, 32);
  const float inv = rsqrtf(qv * (1.f / kH) + 1e-5f);
#pragma unroll
  for (int i = 0; i < 4; ++i) {
    int c = lane * 4 + i;
    xr[c] = (__bf16)((v[i] - mean) * inv * g[c] + bt[c]);
  }
}

// ---------------------------------------------------------------------------
// Fused FFN: out = relu(x @ Wf1^T + b1) @ Wf2^T + b2 for a 16-row tile.
// The 16x2048 relu intermediate lives in LDS (two 1024-wide halves, 32 KB).
// Phase 1 uses 4-way register blocking (A-fragment reuse).
// ---------------------------------------------------------------------------
__global__ __launch_bounds__(256) void ffn_kernel(
    const __bf16* __restrict__ X, const __bf16* __restrict__ W1f,
    const float* __restrict__ b1f, const __bf16* __restrict__ W2f,
    const float* __restrict__ b2f, __bf16* __restrict__ Out) {
  __shared__ __align__(16) __bf16 h1[16][1024];
  const int lane = threadIdx.x & 31, wave = threadIdx.x >> 5;
  const __bf16* Arow = X + (size_t)blockIdx.x * 16 * kH;
  const int row = (lane >> 4) << 3;
  const int cl = lane & 15;
  v8f c2 = {};
  for (int half = 0; half < 2; ++half) {
    for (int grp = 0; grp < 2; ++grp) {
      const int lbase = wave * 128 + grp * 64;   // local col base in h1
      const int nbase = half * 1024 + lbase;     // col base in F
      v8f cc0 = {}, cc1 = {}, cc2 = {}, cc3 = {};
#pragma unroll
      for (int k0 = 0; k0 < kH; k0 += 32) {
        v16bf a = frag_a(Arow, kH, k0);
        cc0 = wmma_bf(a, frag_b(W1f, kH, nbase,      k0), cc0);
        cc1 = wmma_bf(a, frag_b(W1f, kH, nbase + 16, k0), cc1);
        cc2 = wmma_bf(a, frag_b(W1f, kH, nbase + 32, k0), cc2);
        cc3 = wmma_bf(a, frag_b(W1f, kH, nbase + 48, k0), cc3);
      }
      auto emit = [&](v8f c, int nt) {
        const int col = lbase + nt + cl;
        const float bvv = b1f[half * 1024 + col];
#pragma unroll
        for (int i = 0; i < 8; ++i) {
          float v = c[i] + bvv;
          h1[row + i][col] = (__bf16)(v > 0.f ? v : 0.f);
        }
      };
      emit(cc0, 0); emit(cc1, 16); emit(cc2, 32); emit(cc3, 48);
    }
    __syncthreads();
    for (int k0 = 0; k0 < 1024; k0 += 32)
      c2 = wmma_bf(frag_a(&h1[0][0], 1024, k0),
                   frag_b(W2f, kF, wave * 16, half * 1024 + k0), c2);
    __syncthreads();  // before next half overwrites h1
  }
  const int col = wave * 16 + cl;
#pragma unroll
  for (int i = 0; i < 8; ++i)
    Out[((size_t)blockIdx.x * 16 + row + i) * kH + col] = (__bf16)(c2[i] + b2f[col]);
}

// ---------------------------------------------------------------------------
// [B*TT, O] fp32 -> [TT, B, O] fp32 output permute
// ---------------------------------------------------------------------------
__global__ void out_transpose(const float* __restrict__ in, float* __restrict__ out) {
  int i = blockIdx.x * 256 + threadIdx.x;   // over B*TT*O
  int o = i & 31;
  int r = i >> 5;          // = b*TT + t
  int t = r & 127;
  int b = r >> 7;
  out[(size_t)t * (kB * kO) + b * kO + o] = in[i];
}

// ---------------------------------------------------------------------------
// Host orchestration
// ---------------------------------------------------------------------------
extern "C" void kernel_launch(void* const* d_in, const int* in_sizes, int n_in,
                              void* d_out, int out_size, void* d_ws, size_t ws_size,
                              hipStream_t stream) {
  (void)in_sizes; (void)n_in; (void)out_size; (void)ws_size;
  const float* coeffs = (const float*)d_in[0];
  const float* tgt    = (const float*)d_in[1];
  const float* Wi  = (const float*)d_in[2];  const float* bi  = (const float*)d_in[3];
  const float* W1  = (const float*)d_in[4];  const float* b1  = (const float*)d_in[5];
  const float* W2  = (const float*)d_in[6];  const float* b2  = (const float*)d_in[7];
  const float* W3  = (const float*)d_in[8];  const float* b3  = (const float*)d_in[9];
  const float* W4  = (const float*)d_in[10]; const float* b4  = (const float*)d_in[11];
  const float* We  = (const float*)d_in[12]; const float* be  = (const float*)d_in[13];
  const float* Wq  = (const float*)d_in[14]; const float* bq  = (const float*)d_in[15];
  const float* Wk  = (const float*)d_in[16]; const float* bk  = (const float*)d_in[17];
  const float* Wv  = (const float*)d_in[18]; const float* bv  = (const float*)d_in[19];
  const float* Wo  = (const float*)d_in[20]; const float* bo  = (const float*)d_in[21];
  const float* Uq  = (const float*)d_in[22]; const float* bqx = (const float*)d_in[23];
  const float* Uk  = (const float*)d_in[24]; const float* bkx = (const float*)d_in[25];
  const float* Uv  = (const float*)d_in[26]; const float* bvx = (const float*)d_in[27];
  const float* Uo  = (const float*)d_in[28]; const float* box_= (const float*)d_in[29];
  const float* Wf1 = (const float*)d_in[30]; const float* bf1 = (const float*)d_in[31];
  const float* Wf2 = (const float*)d_in[32]; const float* bf2 = (const float*)d_in[33];
  const float* g1  = (const float*)d_in[34]; const float* bt1 = (const float*)d_in[35];
  const float* g2  = (const float*)d_in[36]; const float* bt2 = (const float*)d_in[37];
  const float* g3  = (const float*)d_in[38]; const float* bt3 = (const float*)d_in[39];
  const float* Wfc = (const float*)d_in[40]; const float* bfc = (const float*)d_in[41];

  char* wsb = (char*)d_ws;
  size_t off = 0;
  auto alloc  = [&](size_t bytes) -> char* {
    char* p = wsb + off;
    off += (bytes + 255) & ~(size_t)255;
    return p;
  };
  auto balloc = [&](size_t elems) -> __bf16* { return (__bf16*)alloc(elems * 2); };

  const int MT = kB * kTT;  // 32768 decoder rows
  const int MM = kB * kTX;  // 32512 memory rows

  __bf16* Wi_b  = balloc((size_t)kH * kC);
  __bf16* W1_b  = balloc((size_t)kH * kH);
  __bf16* W2_b  = balloc((size_t)kH * kH);
  __bf16* W3_b  = balloc((size_t)kH * kH);
  __bf16* W4_b  = balloc((size_t)kC * kH * kH);
  __bf16* We_b  = balloc((size_t)kH * kC);
  __bf16* Wq_b  = balloc((size_t)kL * kH * kH);
  __bf16* Wk_b  = balloc((size_t)kL * kH * kH);
  __bf16* Wv_b  = balloc((size_t)kL * kH * kH);
  __bf16* Wo_b  = balloc((size_t)kL * kH * kH);
  __bf16* Uq_b  = balloc((size_t)kL * kH * kH);
  __bf16* Uk_b  = balloc((size_t)kL * kH * kH);
  __bf16* Uv_b  = balloc((size_t)kL * kH * kH);
  __bf16* Uo_b  = balloc((size_t)kL * kH * kH);
  __bf16* Wf1_b = balloc((size_t)kL * kF * kH);
  __bf16* Wf2_b = balloc((size_t)kL * kH * kF);
  __bf16* Wfc_b = balloc((size_t)kO * kH);
  __bf16* tgt_b = balloc((size_t)MT * kC);
  __bf16* mem_b = balloc((size_t)MM * kH);
  __bf16* x_b   = balloc((size_t)MT * kH);
  __bf16* q_b   = balloc((size_t)MT * kH);
  __bf16* k_b   = balloc((size_t)MT * kH);
  __bf16* v_b   = balloc((size_t)MT * kH);
  __bf16* ao_b  = balloc((size_t)MT * kH);
  __bf16* sa_b  = balloc((size_t)MT * kH);
  float*  fc_f  = (float*)alloc((size_t)MT * kO * 4);

  auto cvt = [&](const float* s, __bf16* d, size_t n) {
    cvt_kernel<<<dim3((unsigned)((n + 255) / 256)), 256, 0, stream>>>(s, d, (int)n);
  };
  cvt(Wi, Wi_b, (size_t)kH * kC);
  cvt(W1, W1_b, (size_t)kH * kH);
  cvt(W2, W2_b, (size_t)kH * kH);
  cvt(W3, W3_b, (size_t)kH * kH);
  cvt(W4, W4_b, (size_t)kC * kH * kH);
  cvt(We, We_b, (size_t)kH * kC);
  cvt(Wq, Wq_b, (size_t)kL * kH * kH);
  cvt(Wk, Wk_b, (size_t)kL * kH * kH);
  cvt(Wv, Wv_b, (size_t)kL * kH * kH);
  cvt(Wo, Wo_b, (size_t)kL * kH * kH);
  cvt(Uq, Uq_b, (size_t)kL * kH * kH);
  cvt(Uk, Uk_b, (size_t)kL * kH * kH);
  cvt(Uv, Uv_b, (size_t)kL * kH * kH);
  cvt(Uo, Uo_b, (size_t)kL * kH * kH);
  cvt(Wf1, Wf1_b, (size_t)kL * kF * kH);
  cvt(Wf2, Wf2_b, (size_t)kL * kH * kF);
  cvt(Wfc, Wfc_b, (size_t)kO * kH);
  cvt(tgt, tgt_b, (size_t)MT * kC);

  // Neural CDE: 16 persistent workgroups, 16 batch samples each.
  cde_kernel<<<kB / 16, 256, 0, stream>>>(coeffs, Wi_b, bi, W1_b, b1, W2_b, b2,
                                          W3_b, b3, W4_b, b4, mem_b);

  // Register-blocked N=128 GEMM launcher
  auto gemm128 = [&](const __bf16* A, const __bf16* Wt, const float* bias,
                     __bf16* Cb, int M, int K) {
    gemm128_kernel<<<dim3((unsigned)(M / 64)), 256, 0, stream>>>(A, Wt, bias, Cb, K);
  };

  // decoder embedding
  gemm128(tgt_b, We_b, be, x_b, MT, kC);

  for (int l = 0; l < kL; ++l) {
    const size_t wo = (size_t)l * kH * kH;
    // self-attention
    gemm128(x_b, Wq_b + wo, bq + l * kH, q_b, MT, kH);
    gemm128(x_b, Wk_b + wo, bk + l * kH, k_b, MT, kH);
    gemm128(x_b, Wv_b + wo, bv + l * kH, v_b, MT, kH);
    attn_kernel<<<kB * kNH, 128, 0, stream>>>(q_b, k_b, v_b, ao_b, kTT);
    gemm128(ao_b, Wo_b + wo, bo + l * kH, sa_b, MT, kH);
    ln_kernel<<<MT / 8, 256, 0, stream>>>(x_b, sa_b, g1 + l * kH, bt1 + l * kH);
    // cross-attention against CDE memory
    gemm128(x_b, Uq_b + wo, bqx + l * kH, q_b, MT, kH);
    gemm128(mem_b, Uk_b + wo, bkx + l * kH, k_b, MM, kH);
    gemm128(mem_b, Uv_b + wo, bvx + l * kH, v_b, MM, kH);
    attn_kernel<<<kB * kNH, 128, 0, stream>>>(q_b, k_b, v_b, ao_b, kTX);
    gemm128(ao_b, Uo_b + wo, box_ + l * kH, sa_b, MT, kH);
    ln_kernel<<<MT / 8, 256, 0, stream>>>(x_b, sa_b, g2 + l * kH, bt2 + l * kH);
    // fused FFN
    ffn_kernel<<<MT / 16, 256, 0, stream>>>(x_b, Wf1_b + (size_t)l * kF * kH,
                                            bf1 + l * kF, Wf2_b + (size_t)l * kH * kF,
                                            bf2 + l * kH, sa_b);
    ln_kernel<<<MT / 8, 256, 0, stream>>>(x_b, sa_b, g3 + l * kH, bt3 + l * kH);
  }

  // final projection (N=32) + [TT,B,O] permute
  gemm_kernel<<<dim3(1, (unsigned)(MT / 16)), 64, 0, stream>>>(x_b, Wfc_b, bfc,
                                                               nullptr, fc_f, kO, kH);
  out_transpose<<<(MT * kO) / 256, 256, 0, stream>>>(fc_f, (float*)d_out);
}
